// RecommenderModel_88184268521791
// MI455X (gfx1250) — compile-verified, matrix operand order checked by reading, not run on hardware
//
#include <hip/hip_runtime.h>
#include <math.h>

typedef __attribute__((ext_vector_type(16))) _Float16 v16h;
typedef __attribute__((ext_vector_type(8)))  float    v8f;

#define B_TOTAL 16384
#define LMAX    20
#define MT      64            // batch rows per block
#define K1      160           // padded K for stage-1 GEMM (item 144 -> 160)
#define N1      256
#define K2      256
#define N2      128
#define HSTRIDE 264           // hidden LDS row stride in halves (256 + 8 pad)

// workspace layout in halves (f16): transposed [N][K] weight copies
#define OFF_UW1T 0                         // [256][160]
#define OFF_IW1T (OFF_UW1T + 256*160)      // [256][160], K rows >=144 zeroed
#define OFF_UW2T (OFF_IW1T + 256*160)      // [128][256]
#define OFF_IW2T (OFF_UW2T + 128*256)      // [128][256]
#define WS_HALVES (OFF_IW2T + 128*256)     // 147456 halves = 294912 bytes

// LDS carve (bytes)
#define SM_FEATU 0
#define SM_FEATI (SM_FEATU + MT*K1*2)      // 20480
#define SM_HIDU  (SM_FEATI + MT*K1*2)      // 40960
#define SM_HIDI  (SM_HIDU  + MT*HSTRIDE*2) // 74752
#define SM_ACC   (SM_HIDI  + MT*HSTRIDE*2) // 108544
#define SM_TOTAL (SM_ACC   + MT*4)         // 108800

// ---------------------------------------------------------------------------
// Kernel 0: convert weights to f16 and transpose to [N][K] (B-fragment friendly)
// ---------------------------------------------------------------------------
__global__ void prep_weights(const float* __restrict__ uW1, const float* __restrict__ iW1,
                             const float* __restrict__ uW2, const float* __restrict__ iW2,
                             _Float16* __restrict__ ws) {
  for (int i = blockIdx.x * blockDim.x + threadIdx.x; i < WS_HALVES;
       i += gridDim.x * blockDim.x) {
    if (i < OFF_IW1T) {
      int n = i / K1, k = i % K1;
      ws[i] = (_Float16)uW1[k * 256 + n];                 // uW1: [160][256]
    } else if (i < OFF_UW2T) {
      int j = i - OFF_IW1T; int n = j / K1, k = j % K1;
      ws[i] = (_Float16)((k < 144) ? iW1[k * 256 + n] : 0.0f);  // iW1: [144][256] padded
    } else if (i < OFF_IW2T) {
      int j = i - OFF_UW2T; int n = j / K2, k = j % K2;
      ws[i] = (_Float16)uW2[k * 128 + n];                 // uW2: [256][128]
    } else {
      int j = i - OFF_IW2T; int n = j / K2, k = j % K2;
      ws[i] = (_Float16)iW2[k * 128 + n];                 // iW2: [256][128]
    }
  }
}

// Load a 16-half A/B fragment: elements 0..7 at p[kb..kb+7], 8..15 at p[kb+16..kb+23]
__device__ __forceinline__ v16h load_frag(const _Float16* p) {
  const uint4* q = (const uint4*)p;
  v16h a;
  ((uint4*)&a)[0] = q[0];
  ((uint4*)&a)[1] = q[2];   // +32 bytes
  return a;
}

__device__ __forceinline__ v8f wmma_f16(v16h a, v16h b, v8f c) {
  return __builtin_amdgcn_wmma_f32_16x16x32_f16(false, a, false, b, (short)0, c,
                                                false, false);
}

// ---------------------------------------------------------------------------
// Kernel 1: fused gather -> MLP(user) / MLP(item) -> dot -> sigmoid
// 256 threads (8 wave32), 64 rows per block, 256 blocks
// ---------------------------------------------------------------------------
__global__ __launch_bounds__(256) void fused_rec(
    const int* __restrict__ user_id,    const int* __restrict__ gender_id,
    const int* __restrict__ job_id,     const int* __restrict__ user_city,
    const int* __restrict__ age_bucket, const int* __restrict__ user_labels,
    const int* __restrict__ user_len,   const int* __restrict__ item_id,
    const int* __restrict__ category_id,const int* __restrict__ item_city,
    const int* __restrict__ item_labels,const int* __restrict__ item_len,
    const float* __restrict__ uid_tab,  const float* __restrict__ gen_tab,
    const float* __restrict__ job_tab,  const float* __restrict__ city_tab,
    const float* __restrict__ age_tab,  const float* __restrict__ lab_tab,
    const float* __restrict__ iid_tab,  const float* __restrict__ cat_tab,
    const float* __restrict__ ub1, const float* __restrict__ ub2,
    const float* __restrict__ ib1, const float* __restrict__ ib2,
    const _Float16* __restrict__ ws, float* __restrict__ out) {
  extern __shared__ char smem[];
  _Float16* featU = (_Float16*)(smem + SM_FEATU);   // [64][160] f16
  _Float16* featI = (_Float16*)(smem + SM_FEATI);   // [64][160] f16 (cols 144..159 = 0)
  _Float16* hidU  = (_Float16*)(smem + SM_HIDU);    // [64][264] f16
  _Float16* hidI  = (_Float16*)(smem + SM_HIDI);    // [64][264] f16
  float*    rowAcc= (float*)   (smem + SM_ACC);     // [64] f32

  const int t = threadIdx.x;
  const int rowBase = blockIdx.x * MT;

  if (t < MT) rowAcc[t] = 0.0f;

  // Prefetch the (L2-hot) f16 weight pool toward this WGP.
  {
    const char* wp = (const char*)ws;
    for (int off = t * 128; off < WS_HALVES * 2; off += 256 * 128)
      __builtin_prefetch(wp + off, 0, 1);
  }

  // ---------------- gather: build feature tiles in LDS ----------------
  // user: [0:64) id | [64:80) gender | [80:96) job | [96:112) city | [112:128) age | [128:160) label-mean
  for (int i = t; i < MT * 64; i += 256) {
    int r = i >> 6, d = i & 63;
    featU[r * K1 + d] = (_Float16)uid_tab[(size_t)user_id[rowBase + r] * 64 + d];
  }
  for (int i = t; i < MT * 16; i += 256) {
    int r = i >> 4, d = i & 15;
    featU[r * K1 + 64 + d] = (_Float16)gen_tab[gender_id[rowBase + r] * 16 + d];
  }
  for (int i = t; i < MT * 16; i += 256) {
    int r = i >> 4, d = i & 15;
    featU[r * K1 + 80 + d] = (_Float16)job_tab[job_id[rowBase + r] * 16 + d];
  }
  for (int i = t; i < MT * 16; i += 256) {
    int r = i >> 4, d = i & 15;
    featU[r * K1 + 96 + d] = (_Float16)city_tab[user_city[rowBase + r] * 16 + d];
  }
  for (int i = t; i < MT * 16; i += 256) {
    int r = i >> 4, d = i & 15;
    featU[r * K1 + 112 + d] = (_Float16)age_tab[age_bucket[rowBase + r] * 16 + d];
  }
  for (int i = t; i < MT * 32; i += 256) {       // one row per wave per iter, d = lane
    int r = i >> 5, d = i & 31;
    int row = rowBase + r, len = user_len[row];
    float s = 0.0f;
    for (int j = 0; j < len; ++j)
      s += lab_tab[(size_t)user_labels[row * LMAX + j] * 32 + d];
    featU[r * K1 + 128 + d] = (_Float16)(len > 0 ? s / (float)len : 0.0f);
  }
  // item: [0:64) id | [64:96) category | [96:112) city | [112:144) label-mean | [144:160) zero-pad
  for (int i = t; i < MT * 64; i += 256) {
    int r = i >> 6, d = i & 63;
    featI[r * K1 + d] = (_Float16)iid_tab[(size_t)item_id[rowBase + r] * 64 + d];
  }
  for (int i = t; i < MT * 32; i += 256) {
    int r = i >> 5, d = i & 31;
    featI[r * K1 + 64 + d] = (_Float16)cat_tab[(size_t)category_id[rowBase + r] * 32 + d];
  }
  for (int i = t; i < MT * 16; i += 256) {
    int r = i >> 4, d = i & 15;
    featI[r * K1 + 96 + d] = (_Float16)city_tab[item_city[rowBase + r] * 16 + d];
  }
  for (int i = t; i < MT * 32; i += 256) {
    int r = i >> 5, d = i & 31;
    int row = rowBase + r, len = item_len[row];
    float s = 0.0f;
    for (int j = 0; j < len; ++j)
      s += lab_tab[(size_t)item_labels[row * LMAX + j] * 32 + d];
    featI[r * K1 + 112 + d] = (_Float16)(len > 0 ? s / (float)len : 0.0f);
  }
  for (int i = t; i < MT * 16; i += 256) {
    int r = i >> 4, d = i & 15;
    featI[r * K1 + 144 + d] = (_Float16)0.0f;
  }
  __syncthreads();

  const int wave = t >> 5, lane = t & 31;
  const int lrow = lane & 15, lhalf = lane >> 4;

  // ---------------- GEMM1: hid = act(feat @ W1 + b1), M=64 N=256 K=160 -------
  // 2 paths x 4 m-tiles x 16 n-tiles = 128 tile jobs; 16 per wave (uniform).
  for (int jj = 0; jj < 16; ++jj) {
    int job  = wave * 16 + jj;
    int path = job >> 6;                 // 0 = user, 1 = item
    int rem  = job & 63;
    int mt = rem & 3, nt = rem >> 2;
    const _Float16* feat = path ? featI : featU;
    const _Float16* w1   = ws + (path ? OFF_IW1T : OFF_UW1T);
    const float*    b1   = path ? ib1 : ub1;
    _Float16*       hid  = path ? hidI : hidU;

    v8f acc = {};
#pragma unroll
    for (int ks = 0; ks < 5; ++ks) {
      int kb = ks * 32 + lhalf * 8;
      v16h a = load_frag(&feat[(mt * 16 + lrow) * K1 + kb]);
      v16h b = load_frag(&w1[(nt * 16 + lrow) * K1 + kb]);
      acc = wmma_f16(a, b, acc);
    }
    float bias = b1[nt * 16 + lrow];
#pragma unroll
    for (int i = 0; i < 8; ++i) {
      float v = acc[i] + bias;
      if (path == 0) v = v > 0.0f ? v : 0.0f;   // ReLU only on user path
      hid[(mt * 16 + i + 8 * lhalf) * HSTRIDE + nt * 16 + lrow] = (_Float16)v;
    }
  }
  __syncthreads();

  // ---------------- GEMM2 + fused dot: M=64 N=128 K=256 ---------------------
  // 4 m-tiles x 8 n-tiles = 32 tile-pairs; 4 per wave (uniform).
  {
    int mt = wave & 3;
    int ntbase = (wave >> 2) * 4;
    for (int j = 0; j < 4; ++j) {
      int nt = ntbase + j;
      v8f aU = {}, aI = {};
#pragma unroll
      for (int ks = 0; ks < 8; ++ks) {
        int kb = ks * 32 + lhalf * 8;
        v16h au = load_frag(&hidU[(mt * 16 + lrow) * HSTRIDE + kb]);
        v16h bu = load_frag(&ws[OFF_UW2T + (nt * 16 + lrow) * K2 + kb]);
        aU = wmma_f16(au, bu, aU);
        v16h ai = load_frag(&hidI[(mt * 16 + lrow) * HSTRIDE + kb]);
        v16h bi = load_frag(&ws[OFF_IW2T + (nt * 16 + lrow) * K2 + kb]);
        aI = wmma_f16(ai, bi, aI);
      }
      float bu2 = ub2[nt * 16 + lrow];
      float bi2 = ib2[nt * 16 + lrow];
      float p[8];
#pragma unroll
      for (int i = 0; i < 8; ++i) p[i] = (aU[i] + bu2) * (aI[i] + bi2);
      // reduce across the 16 columns held by each lane-half
#pragma unroll
      for (int m = 1; m < 16; m <<= 1) {
#pragma unroll
        for (int i = 0; i < 8; ++i) p[i] += __shfl_xor(p[i], m, 32);
      }
      if (lrow == 0) {
#pragma unroll
        for (int i = 0; i < 8; ++i)
          atomicAdd(&rowAcc[mt * 16 + i + 8 * lhalf], p[i]);  // ds_add_f32
      }
    }
  }
  __syncthreads();

  if (t < MT) {
    float x = rowAcc[t];
    out[rowBase + t] = 1.0f / (1.0f + expf(-x));
  }
}

// ---------------------------------------------------------------------------
extern "C" void kernel_launch(void* const* d_in, const int* in_sizes, int n_in,
                              void* d_out, int out_size, void* d_ws, size_t ws_size,
                              hipStream_t stream) {
  (void)in_sizes; (void)n_in; (void)out_size; (void)ws_size;
  const int* user_id     = (const int*)d_in[0];
  const int* gender_id   = (const int*)d_in[1];
  const int* job_id      = (const int*)d_in[2];
  const int* user_city   = (const int*)d_in[3];
  const int* age_bucket  = (const int*)d_in[4];
  const int* user_labels = (const int*)d_in[5];
  const int* user_len    = (const int*)d_in[6];
  const int* item_id     = (const int*)d_in[7];
  const int* category_id = (const int*)d_in[8];
  const int* item_city   = (const int*)d_in[9];
  const int* item_labels = (const int*)d_in[10];
  const int* item_len    = (const int*)d_in[11];
  const float* uid_tab = (const float*)d_in[12];
  const float* gen_tab = (const float*)d_in[13];
  const float* job_tab = (const float*)d_in[14];
  const float* city_tab= (const float*)d_in[15];
  const float* age_tab = (const float*)d_in[16];
  const float* lab_tab = (const float*)d_in[17];
  const float* iid_tab = (const float*)d_in[18];
  const float* cat_tab = (const float*)d_in[19];
  const float* uW1 = (const float*)d_in[20];
  const float* ub1 = (const float*)d_in[21];
  const float* uW2 = (const float*)d_in[22];
  const float* ub2 = (const float*)d_in[23];
  const float* iW1 = (const float*)d_in[24];
  const float* ib1 = (const float*)d_in[25];
  const float* iW2 = (const float*)d_in[26];
  const float* ib2 = (const float*)d_in[27];

  _Float16* wsh = (_Float16*)d_ws;   // needs 294,912 bytes

  prep_weights<<<(WS_HALVES + 255) / 256, 256, 0, stream>>>(uW1, iW1, uW2, iW2, wsh);

  fused_rec<<<B_TOTAL / MT, 256, SM_TOTAL, stream>>>(
      user_id, gender_id, job_id, user_city, age_bucket, user_labels, user_len,
      item_id, category_id, item_city, item_labels, item_len,
      uid_tab, gen_tab, job_tab, city_tab, age_tab, lab_tab, iid_tab, cat_tab,
      ub1, ub2, ib1, ib2, (const _Float16*)wsh, (float*)d_out);
}
